// HierarchicalFeatureExtractor_72172630442457
// MI455X (gfx1250) — compile-verified, matrix operand order checked by reading, not run on hardware
//
#include <hip/hip_runtime.h>
#include <hip/hip_bf16.h>
#include <math.h>

typedef _Float16 half_t;
typedef __attribute__((ext_vector_type(16))) _Float16 v16h;
typedef __attribute__((ext_vector_type(8)))  float    v8f;
typedef __attribute__((ext_vector_type(4)))  unsigned int u32x4;
typedef __attribute__((ext_vector_type(8)))  int      i32x8;
typedef __attribute__((ext_vector_type(4)))  int      i32x4;

#define BB 32
#define HH 56
#define WW 56
#define CC 256
#define WS 7
#define SHIFT 3
#define HEADS 8
#define HD 32
#define NN 49
#define NWTOT 2048            // BB * 64 windows
#define TT 100352             // BB*HH*WW tokens
#define C4 1024

static __device__ __forceinline__ int wave_id() { return threadIdx.x >> 5; }
static __device__ __forceinline__ int lane_id() { return threadIdx.x & 31; }

// ---- WMMA fragment loaders (gfx1250 wave32 layouts, CDNA5 ISA) ----
// A matrix 16x32 f16, row-major source a[m*lda + k]
static __device__ __forceinline__ v16h load_frag_a(const half_t* a, int lda) {
  int l = lane_id();
  int m  = l & 15;
  int kb = (l >> 4) * 8;
  v16h f;
#pragma unroll
  for (int v = 0; v < 8; ++v) {
    int k = ((v >> 2) << 4) + kb + ((v & 3) << 1);
    f[2 * v]     = a[m * lda + k];
    f[2 * v + 1] = a[m * lda + k + 1];
  }
  return f;
}
// B matrix 32x16 f16 from row-major b[k*ldb + n]  (B(k,n) = b[k][n])
static __device__ __forceinline__ v16h load_frag_b(const half_t* b, int ldb) {
  int l = lane_id();
  int n  = l & 15;
  int kb = (l >> 4) * 8;
  v16h f;
#pragma unroll
  for (int v = 0; v < 8; ++v) {
    int k = ((v >> 2) << 4) + kb + ((v & 3) << 1);
    f[2 * v]     = b[k * ldb + n];
    f[2 * v + 1] = b[(k + 1) * ldb + n];
  }
  return f;
}
// B matrix 32x16 where B(k,n) = w[n*ldk + k]  (i.e. W^T with W row-major [out,in])
static __device__ __forceinline__ v16h load_frag_bT(const half_t* w, int ldk) {
  int l = lane_id();
  int n  = l & 15;
  int kb = (l >> 4) * 8;
  v16h f;
#pragma unroll
  for (int v = 0; v < 8; ++v) {
    int k = ((v >> 2) << 4) + kb + ((v & 3) << 1);
    f[2 * v]     = w[n * ldk + k];
    f[2 * v + 1] = w[n * ldk + k + 1];
  }
  return f;
}
static __device__ __forceinline__ v8f wmma_f16(v16h a, v16h b, v8f c) {
  return __builtin_amdgcn_wmma_f32_16x16x32_f16(false, a, false, b, (short)0, c, false, false);
}

// ---- Tensor Data Mover row-block fetch (CDNA5 ISA) ----
// Loads `rows` rows of `row_units` 8-byte units from global (row stride
// `stride_units` units) into LDS at byte offset lds_off. pad_code selects the
// LDS pad interval (5 -> every 256B, 6 -> every 512B); pad_amount=3 -> 16B,
// giving 272B / 528B LDS row pitch (bank-conflict-free fragment reads).
#if __has_builtin(__builtin_amdgcn_tensor_load_to_lds)
#define HAVE_TDM 1
static __device__ __forceinline__ void tdm_load_rows(const half_t* src, unsigned lds_off,
                                                     unsigned row_units, unsigned rows,
                                                     unsigned stride_units, int pad_code) {
  unsigned long long ga = (unsigned long long)(size_t)src;
  u32x4 g0;
  g0[0] = 1u;                                   // count=1, is_restore=0, gather=0
  g0[1] = lds_off;                              // lds_addr (bytes)
  g0[2] = (unsigned)ga;                         // global_addr[31:0]
  g0[3] = (unsigned)((ga >> 32) & 0x01FFFFFFu) | 0x80000000u;  // ga[56:32] | type=2
  i32x8 g1;
  g1[0] = (3 << 16) | (1 << 20) | (pad_code << 22) | (3 << 25); // 8B units, pad on
  g1[1] = (int)(row_units << 16);               // tensor_dim0[15:0]
  g1[2] = (int)(rows << 16);                    // tensor_dim1[15:0]
  g1[3] = (int)(row_units << 16);               // tile_dim0
  g1[4] = (int)rows;                            // tile_dim1 (tile_dim2=0)
  g1[5] = (int)stride_units;                    // tensor_dim0_stride[31:0]
  g1[6] = 0;
  g1[7] = 0;
  i32x4 g2 = {0, 0, 0, 0};
  i32x4 g3 = {0, 0, 0, 0};
#if defined(__clang_major__) && (__clang_major__ >= 23)
  i32x8 g4 = {0, 0, 0, 0, 0, 0, 0, 0};
  __builtin_amdgcn_tensor_load_to_lds(g0, g1, g2, g3, g4, 0);
#else
  __builtin_amdgcn_tensor_load_to_lds(g0, g1, g2, g3, 0);
#endif
}
#else
#define HAVE_TDM 0
#endif

// Per-wave 16x256-half slab fetch into a 64x264 tile (no block barrier needed:
// each wave only ever reads its own 16 rows of the A tile).
static __device__ __forceinline__ void wave_load_slab256(half_t (*As)[264],
                                                         const half_t* src, int src_ld) {
  int wv = wave_id();
#if HAVE_TDM
  unsigned lds0 = (unsigned)(size_t)(&As[0][0]) + (unsigned)(wv * 16) * 528u;
  tdm_load_rows(src + (size_t)(wv * 16) * src_ld, lds0, 64u, 16u,
                (unsigned)(src_ld >> 2), 6);
  __builtin_amdgcn_s_wait_tensorcnt(0);
#else
  int l = lane_id();
  for (int idx = l; idx < 16 * 256; idx += 32) {
    int r = idx >> 8, c = idx & 255;
    As[wv * 16 + r][c] = src[(size_t)(wv * 16 + r) * src_ld + c];
  }
#endif
}

// ---------------- weight convert f32 -> f16 ----------------
__global__ void k_cvt(const float* __restrict__ src, half_t* __restrict__ dst, int n) {
  int i = blockIdx.x * 256 + threadIdx.x;
  if (i < n) dst[i] = (half_t)src[i];
}

// ---------------- LayerNorm: one wave per token, C=256 -> 8 per lane -------
__global__ void __launch_bounds__(256) k_ln(const float* __restrict__ x,
                                            const float* __restrict__ g,
                                            const float* __restrict__ bta,
                                            half_t* __restrict__ out) {
  int wv = wave_id(), l = lane_id();
  size_t tok = (size_t)blockIdx.x * 8 + wv;
  if (tok >= (size_t)TT) return;
  const float* row = x + tok * CC;
  float v[8];
  float s = 0.f;
#pragma unroll
  for (int j = 0; j < 8; ++j) { v[j] = row[l + j * 32]; s += v[j]; }
#pragma unroll
  for (int off = 16; off > 0; off >>= 1) s += __shfl_xor(s, off, 32);
  float mean = s * (1.f / 256.f);
  float vs = 0.f;
#pragma unroll
  for (int j = 0; j < 8; ++j) { float d = v[j] - mean; vs += d * d; }
#pragma unroll
  for (int off = 16; off > 0; off >>= 1) vs += __shfl_xor(vs, off, 32);
  float inv = rsqrtf(vs * (1.f / 256.f) + 1e-5f);
#pragma unroll
  for (int j = 0; j < 8; ++j) {
    int c = l + j * 32;
    out[tok * CC + c] = (half_t)((v[j] - mean) * inv * g[c] + bta[c]);
  }
}

// ---------------- depthwise 3x3 conv (SAME, zero pad) ----------------------
__global__ void k_dw(const half_t* __restrict__ xn, const float* __restrict__ w,
                     const float* __restrict__ bias, float* __restrict__ out) {
  size_t idx = (size_t)blockIdx.x * 256 + threadIdx.x;
  if (idx >= (size_t)TT * CC) return;
  int c = (int)(idx & 255);
  size_t tok = idx >> 8;
  int b  = (int)(tok / (HH * WW));
  int hw = (int)(tok % (HH * WW));
  int h = hw / WW, wc = hw % WW;
  float acc = bias[c];
#pragma unroll
  for (int dy = -1; dy <= 1; ++dy)
#pragma unroll
    for (int dx = -1; dx <= 1; ++dx) {
      int hh = h + dy, ww2 = wc + dx;
      if (hh >= 0 && hh < HH && ww2 >= 0 && ww2 < WW)
        acc += (float)xn[((size_t)(b * HH * WW + hh * WW + ww2)) * CC + c] *
               w[((dy + 1) * 3 + (dx + 1)) * CC + c];
    }
  out[idx] = acc;
}

// ---------------- QKV GEMM per window: [49,256] x [256,768]^T --------------
__global__ void __launch_bounds__(128) k_qkv(const half_t* __restrict__ xn,
                                             const half_t* __restrict__ wq,
                                             const float* __restrict__ qkvb,
                                             half_t* __restrict__ qb,
                                             half_t* __restrict__ kb,
                                             half_t* __restrict__ vb) {
  __shared__ alignas(16) half_t As[64][264];
  int wb = blockIdx.x;
  int b = wb >> 6, wi = wb & 63, whi = wi >> 3, wwi = wi & 7;
  for (int idx = threadIdx.x; idx < 64 * 256; idx += 128) {
    int r = idx >> 8, c = idx & 255;
    half_t v = (half_t)0.f;
    if (r < NN) {
      int i = r / WS, j = r % WS;
      int h = (whi * WS + i + SHIFT) % HH;      // roll(x, -SHIFT)
      int w = (wwi * WS + j + SHIFT) % WW;
      v = xn[((size_t)(b * HH * WW + h * WW + w)) * CC + c];
    }
    As[r][c] = v;
  }
  __syncthreads();
  int wv = wave_id(), l = lane_id();
  int r0 = wv * 16, nn = l & 15, rb = (l >> 4) * 8;
  v16h af[8];
#pragma unroll
  for (int ks = 0; ks < 8; ++ks) af[ks] = load_frag_a(&As[r0][ks * 32], 264);
  for (int ot = 0; ot < 48; ++ot) {
    v8f acc = {};
#pragma unroll
    for (int ks = 0; ks < 8; ++ks) {
      v16h bf = load_frag_bT(wq + (size_t)(ot * 16) * CC + ks * 32, CC);
      acc = wmma_f16(af[ks], bf, acc);
    }
    int o = ot * 16 + nn;
    float bias = qkvb[o];
    int part = o >> 8;                   // 0=q 1=k 2=v
    int oc = o & 255;
    int head = oc >> 5, d = oc & 31;
    half_t* dst = part == 0 ? qb : (part == 1 ? kb : vb);
    float scale = (part == 0) ? 0.17677669529663688f : 1.0f;   // hd^-0.5 folded into q
    size_t base = (((size_t)wb * HEADS + head) * 64) * HD + d;
#pragma unroll
    for (int j = 0; j < 8; ++j) {
      int row = r0 + rb + j;
      dst[base + (size_t)row * HD] = (half_t)((acc[j] + bias) * scale);
    }
  }
}

// -------- attention per (window, head): S=q k^T, bias+mask+softmax, P v ----
__global__ void __launch_bounds__(128) k_attn(const half_t* __restrict__ qb,
                                              const half_t* __restrict__ kbuf,
                                              const half_t* __restrict__ vb,
                                              const float* __restrict__ rpb,
                                              half_t* __restrict__ aout) {
  __shared__ float  S[64][65];
  __shared__ half_t P[64][72];
  int gid = blockIdx.x;
  int wb = gid >> 3, head = gid & 7;
  int wi = wb & 63, whi = wi >> 3, wwi = wi & 7;
  const half_t* q = qb   + ((size_t)wb * HEADS + head) * 64 * HD;
  const half_t* k = kbuf + ((size_t)wb * HEADS + head) * 64 * HD;
  const half_t* v = vb   + ((size_t)wb * HEADS + head) * 64 * HD;
  int wv = wave_id(), l = lane_id();
  int r0 = wv * 16, nn = l & 15, rb = (l >> 4) * 8;

  v16h aq = load_frag_a(q + (size_t)r0 * HD, HD);
#pragma unroll
  for (int ct = 0; ct < 4; ++ct) {
    v16h bk = load_frag_bT(k + (size_t)(ct * 16) * HD, HD);   // B(d,m)=k[m][d]
    v8f acc = {};
    acc = wmma_f16(aq, bk, acc);
#pragma unroll
    for (int j = 0; j < 8; ++j) S[r0 + rb + j][ct * 16 + nn] = acc[j];
  }
  __syncthreads();

  int t = threadIdx.x;
  if (t < 64) {
    if (t < NN) {
      int i1 = t / WS, j1 = t % WS;
      int h1 = whi * WS + i1, w1 = wwi * WS + j1;
      int r1 = (h1 < 49 ? 0 : (h1 < 53 ? 1 : 2)) * 3 + (w1 < 49 ? 0 : (w1 < 53 ? 1 : 2));
      float mx = -1e30f;
      for (int m = 0; m < NN; ++m) {
        int i2 = m / WS, j2 = m % WS;
        int h2 = whi * WS + i2, w2 = wwi * WS + j2;
        int r2 = (h2 < 49 ? 0 : (h2 < 53 ? 1 : 2)) * 3 + (w2 < 49 ? 0 : (w2 < 53 ? 1 : 2));
        int ridx = (i1 - i2 + 6) * 13 + (j1 - j2 + 6);
        float s = S[t][m] + rpb[ridx * HEADS + head] + ((r1 != r2) ? -100.f : 0.f);
        S[t][m] = s;
        mx = fmaxf(mx, s);
      }
      float sum = 0.f;
      for (int m = 0; m < NN; ++m) { float e = __expf(S[t][m] - mx); S[t][m] = e; sum += e; }
      float inv = 1.f / sum;
      for (int m = 0; m < NN; ++m) P[t][m] = (half_t)(S[t][m] * inv);
      for (int m = NN; m < 64; ++m) P[t][m] = (half_t)0.f;
    } else {
      for (int m = 0; m < 64; ++m) P[t][m] = (half_t)0.f;
    }
  }
  __syncthreads();

  v16h pa0 = load_frag_a(&P[r0][0], 72);
  v16h pa1 = load_frag_a(&P[r0][32], 72);
#pragma unroll
  for (int ct = 0; ct < 2; ++ct) {
    v8f acc = {};
    v16h b0 = load_frag_b(v + ct * 16, HD);                 // K rows 0..31
    acc = wmma_f16(pa0, b0, acc);
    v16h b1 = load_frag_b(v + 32 * HD + ct * 16, HD);       // K rows 32..63
    acc = wmma_f16(pa1, b1, acc);
    int col = head * HD + ct * 16 + nn;
#pragma unroll
    for (int j = 0; j < 8; ++j) {
      int row = r0 + rb + j;
      aout[((size_t)wb * 64 + row) * CC + col] = (half_t)acc[j];
    }
  }
}

// -------- proj GEMM per window + reverse shift scatter + residual ----------
__global__ void __launch_bounds__(128) k_proj(const half_t* __restrict__ aout,
                                              const half_t* __restrict__ wp,
                                              const float* __restrict__ pb,
                                              const float* __restrict__ x0,
                                              const float* __restrict__ xloc,
                                              float* __restrict__ xmid) {
  __shared__ alignas(16) half_t As[64][264];
  int wb = blockIdx.x;
  int b = wb >> 6, wi = wb & 63, whi = wi >> 3, wwi = wi & 7;
  wave_load_slab256(As, aout + (size_t)wb * 64 * CC, CC);  // per-wave TDM slab
  int wv = wave_id(), l = lane_id();
  int r0 = wv * 16, nn = l & 15, rb = (l >> 4) * 8;
  v16h af[8];
#pragma unroll
  for (int ks = 0; ks < 8; ++ks) af[ks] = load_frag_a(&As[r0][ks * 32], 264);
  for (int ct = 0; ct < 16; ++ct) {
    v8f acc = {};
#pragma unroll
    for (int ks = 0; ks < 8; ++ks) {
      v16h bf = load_frag_bT(wp + (size_t)(ct * 16) * CC + ks * 32, CC);
      acc = wmma_f16(af[ks], bf, acc);
    }
    int c = ct * 16 + nn;
    float bias = pb[c];
#pragma unroll
    for (int j = 0; j < 8; ++j) {
      int r = r0 + rb + j;
      if (r < NN) {
        int i = r / WS, j2 = r % WS;
        int h = (whi * WS + i + SHIFT) % HH;
        int w = (wwi * WS + j2 + SHIFT) % WW;
        size_t tix = (size_t)(b * HH * WW + h * WW + w);
        xmid[tix * CC + c] = x0[tix * CC + c] + acc[j] + bias + 0.1f * xloc[tix * CC + c];
      }
    }
  }
}

// -------- MLP layer 1: [T,256] x W1^T -> gelu -> h1 [T,1024] ---------------
__global__ void __launch_bounds__(128) k_mlp1(const half_t* __restrict__ xin,
                                              const half_t* __restrict__ w1,
                                              const float* __restrict__ b1,
                                              half_t* __restrict__ h1) {
  __shared__ alignas(16) half_t As[64][264];
  size_t row0 = (size_t)blockIdx.x * 64;
  int col0 = blockIdx.y * 64;
  wave_load_slab256(As, xin + row0 * CC, CC);              // per-wave TDM slab
  int wv = wave_id(), l = lane_id();
  int r0 = wv * 16, nn = l & 15, rb = (l >> 4) * 8;
  v16h af[8];
#pragma unroll
  for (int ks = 0; ks < 8; ++ks) af[ks] = load_frag_a(&As[r0][ks * 32], 264);
  for (int ct = 0; ct < 4; ++ct) {
    v8f acc = {};
    int oc0 = col0 + ct * 16;
#pragma unroll
    for (int ks = 0; ks < 8; ++ks) {
      v16h bf = load_frag_bT(w1 + (size_t)oc0 * CC + ks * 32, CC);
      acc = wmma_f16(af[ks], bf, acc);
    }
    int c = oc0 + nn;
    float bias = b1[c];
#pragma unroll
    for (int j = 0; j < 8; ++j) {
      float xv = acc[j] + bias;
      float g = 0.5f * xv * (1.0f + erff(xv * 0.70710678118654752f));
      h1[(row0 + r0 + rb + j) * C4 + c] = (half_t)g;
    }
  }
}

// ---- MLP layers 2/3 share a double-buffered per-wave TDM pipeline over K ----
// K=1024 split into 8 chunks of 128 halfs (256B rows -> pad_code 5, 272B pitch).
// Per wave: issue DMA for chunk kc+1, s_wait_tensorcnt<=1 (in-order completion
// guarantees chunk kc resident), compute chunk kc. No block barriers: each wave
// owns its 16-row slab in both buffers.
#define MLP_K_PIPE(SRC)                                                          \
  const half_t* srcw = (SRC) + (row0 + (size_t)r0) * C4;                         \
  v8f acc[4] = {};                                                               \
  _Pragma("nounroll")                                                            \
  for (int kc = 0; kc < 8; ++kc) {                                               \
    PIPE_FETCH(kc)                                                               \
    v16h af[4];                                                                  \
    _Pragma("unroll")                                                            \
    for (int ks = 0; ks < 4; ++ks) af[ks] = load_frag_a(&As[kc & 1][r0][ks * 32], 136);

#if HAVE_TDM
#define PIPE_PROLOG()                                                            \
  unsigned lds0 = (unsigned)(size_t)(&As[0][0][0]) + (unsigned)(r0 * 272);       \
  tdm_load_rows(srcw0, lds0, 32u, 16u, 256u, 5);
#define PIPE_FETCH(kc)                                                           \
  if ((kc) < 7) {                                                                \
    tdm_load_rows(srcw + ((kc) + 1) * 128,                                       \
                  lds0 + (unsigned)((((kc) + 1) & 1) * 17408), 32u, 16u, 256u, 5); \
    __builtin_amdgcn_s_wait_tensorcnt(1);                                        \
  } else {                                                                       \
    __builtin_amdgcn_s_wait_tensorcnt(0);                                        \
  }
#else
#define PIPE_PROLOG()
#define PIPE_FETCH(kc)                                                           \
  for (int idx = l; idx < 16 * 128; idx += 32) {                                 \
    int r = idx >> 7, cx = idx & 127;                                            \
    As[(kc) & 1][r0 + r][cx] = srcw[(size_t)r * C4 + (kc) * 128 + cx];           \
  }
#endif

// -------- MLP layer 2: [T,1024] x W2^T -> gelu -> h2 [T,1024] --------------
__global__ void __launch_bounds__(128) k_mlp2(const half_t* __restrict__ h1,
                                              const half_t* __restrict__ w2,
                                              const float* __restrict__ b2,
                                              half_t* __restrict__ h2) {
  __shared__ alignas(16) half_t As[2][64][136];
  size_t row0 = (size_t)blockIdx.x * 64;
  int col0 = blockIdx.y * 64;
  int wv = wave_id(), l = lane_id();
  int r0 = wv * 16, nn = l & 15, rb = (l >> 4) * 8;
  const half_t* srcw0 = h1 + (row0 + (size_t)r0) * C4;
  PIPE_PROLOG()
  MLP_K_PIPE(h1)
#pragma unroll
    for (int ct = 0; ct < 4; ++ct) {
#pragma unroll
      for (int ks = 0; ks < 4; ++ks) {
        v16h bf = load_frag_bT(w2 + (size_t)(col0 + ct * 16) * C4 + kc * 128 + ks * 32, C4);
        acc[ct] = wmma_f16(af[ks], bf, acc[ct]);
      }
    }
  }
#pragma unroll
  for (int ct = 0; ct < 4; ++ct) {
    int c = col0 + ct * 16 + nn;
    float bias = b2[c];
#pragma unroll
    for (int j = 0; j < 8; ++j) {
      float xv = acc[ct][j] + bias;
      float g = 0.5f * xv * (1.0f + erff(xv * 0.70710678118654752f));
      h2[(row0 + r0 + rb + j) * C4 + c] = (half_t)g;
    }
  }
}

// -------- MLP layer 3 + residual: out = xmid + h2 x W3^T + b3 --------------
__global__ void __launch_bounds__(128) k_mlp3(const half_t* __restrict__ h2,
                                              const half_t* __restrict__ w3,
                                              const float* __restrict__ b3,
                                              const float* __restrict__ xmid,
                                              float* __restrict__ out) {
  __shared__ alignas(16) half_t As[2][64][136];
  size_t row0 = (size_t)blockIdx.x * 64;
  int col0 = blockIdx.y * 64;
  int wv = wave_id(), l = lane_id();
  int r0 = wv * 16, nn = l & 15, rb = (l >> 4) * 8;
  const half_t* srcw0 = h2 + (row0 + (size_t)r0) * C4;
  PIPE_PROLOG()
  MLP_K_PIPE(h2)
#pragma unroll
    for (int ct = 0; ct < 4; ++ct) {
#pragma unroll
      for (int ks = 0; ks < 4; ++ks) {
        v16h bf = load_frag_bT(w3 + (size_t)(col0 + ct * 16) * C4 + kc * 128 + ks * 32, C4);
        acc[ct] = wmma_f16(af[ks], bf, acc[ct]);
      }
    }
  }
#pragma unroll
  for (int ct = 0; ct < 4; ++ct) {
    int c = col0 + ct * 16 + nn;
    float bias = b3[c];
#pragma unroll
    for (int j = 0; j < 8; ++j) {
      size_t row = row0 + r0 + rb + j;
      out[row * CC + c] = xmid[row * CC + c] + acc[ct][j] + bias;
    }
  }
}

extern "C" void kernel_launch(void* const* d_in, const int* in_sizes, int n_in,
                              void* d_out, int out_size, void* d_ws, size_t ws_size,
                              hipStream_t stream) {
  (void)in_sizes; (void)n_in; (void)out_size; (void)ws_size;
  const float* x      = (const float*)d_in[0];
  const float* ln1_g  = (const float*)d_in[1];
  const float* ln1_b  = (const float*)d_in[2];
  const float* dw_w   = (const float*)d_in[3];
  const float* dw_b   = (const float*)d_in[4];
  const float* qkv_w  = (const float*)d_in[5];
  const float* qkv_b  = (const float*)d_in[6];
  const float* rpb    = (const float*)d_in[7];
  const float* proj_w = (const float*)d_in[8];
  const float* proj_b = (const float*)d_in[9];
  const float* ln2_g  = (const float*)d_in[10];
  const float* ln2_b  = (const float*)d_in[11];
  const float* w1     = (const float*)d_in[12];
  const float* b1     = (const float*)d_in[13];
  const float* w2     = (const float*)d_in[14];
  const float* b2     = (const float*)d_in[15];
  const float* w3     = (const float*)d_in[16];
  const float* b3     = (const float*)d_in[17];

  char* ws = (char*)d_ws;
  size_t off = 0;
  auto alloc = [&](size_t bytes) -> char* {
    char* p = ws + off;
    off += (bytes + 255) & ~(size_t)255;
    return p;
  };
  const size_t TC = (size_t)TT * CC;

  half_t* qkv_w_h  = (half_t*)alloc((size_t)768 * 256 * sizeof(half_t));
  half_t* proj_w_h = (half_t*)alloc((size_t)256 * 256 * sizeof(half_t));
  half_t* w1_h     = (half_t*)alloc((size_t)1024 * 256 * sizeof(half_t));
  half_t* w2_h     = (half_t*)alloc((size_t)1024 * 1024 * sizeof(half_t));
  half_t* w3_h     = (half_t*)alloc((size_t)256 * 1024 * sizeof(half_t));
  half_t* xn_h     = (half_t*)alloc(TC * sizeof(half_t));
  float*  xloc     = (float*) alloc(TC * sizeof(float));
  half_t* qb       = (half_t*)alloc((size_t)NWTOT * HEADS * 64 * HD * sizeof(half_t));
  half_t* kb       = (half_t*)alloc((size_t)NWTOT * HEADS * 64 * HD * sizeof(half_t));
  half_t* vb       = (half_t*)alloc((size_t)NWTOT * HEADS * 64 * HD * sizeof(half_t));
  half_t* aout     = (half_t*)alloc((size_t)NWTOT * 64 * CC * sizeof(half_t));
  float*  xmid     = (float*) alloc(TC * sizeof(float));
  half_t* h2       = (half_t*)alloc((size_t)TT * C4 * sizeof(half_t));
  half_t* xn2 = xn_h;          // alias: xn dead after QKV/dwconv, LN2 runs later
  half_t* h1  = qb;            // alias: q/k/v + aout (256MB) dead before MLP1 (205.5MB)

  float* out = (float*)d_out;

  k_cvt<<<(768 * 256 + 255) / 256, 256, 0, stream>>>(qkv_w, qkv_w_h, 768 * 256);
  k_cvt<<<(256 * 256 + 255) / 256, 256, 0, stream>>>(proj_w, proj_w_h, 256 * 256);
  k_cvt<<<(1024 * 256 + 255) / 256, 256, 0, stream>>>(w1, w1_h, 1024 * 256);
  k_cvt<<<(1024 * 1024 + 255) / 256, 256, 0, stream>>>(w2, w2_h, 1024 * 1024);
  k_cvt<<<(256 * 1024 + 255) / 256, 256, 0, stream>>>(w3, w3_h, 256 * 1024);

  k_ln<<<TT / 8, 256, 0, stream>>>(x, ln1_g, ln1_b, xn_h);
  k_dw<<<(int)((TC + 255) / 256), 256, 0, stream>>>(xn_h, dw_w, dw_b, xloc);
  k_qkv<<<NWTOT, 128, 0, stream>>>(xn_h, qkv_w_h, qkv_b, qb, kb, vb);
  k_attn<<<NWTOT * HEADS, 128, 0, stream>>>(qb, kb, vb, rpb, aout);
  k_proj<<<NWTOT, 128, 0, stream>>>(aout, proj_w_h, proj_b, x, xloc, xmid);
  k_ln<<<TT / 8, 256, 0, stream>>>(xmid, ln2_g, ln2_b, xn2);
  k_mlp1<<<dim3(TT / 64, 16), 128, 0, stream>>>(xn2, w1_h, b1, h1);
  k_mlp2<<<dim3(TT / 64, 16), 128, 0, stream>>>(h1, w2_h, b2, h2);
  k_mlp3<<<dim3(TT / 64, 4), 128, 0, stream>>>(h2, w3_h, b3, xmid, out);
}
